// GNN_29695403884656
// MI455X (gfx1250) — matrix operand order repaired and checked by reading, not compiled, not run on hardware
//
#include <hip/hip_runtime.h>
#include <math.h>

#define NN     100000
#define EE     500000
#define INDIM  256
#define NHID   128
#define TT     3
#define RR     6
#define HH     8
#define LL     2
#define DKK    16
#define MAXLEN 240
// node-permutation capacity: each type bucket padded to a 128-row boundary
#define PAD_CAP (((NN + TT * 128 + 127) / 128) * 128)

typedef __attribute__((ext_vector_type(16))) __bf16 v16bf;
typedef __attribute__((ext_vector_type(8)))  float  v8f;

union Frag { v16bf v; unsigned int u[8]; uint4 q[2]; };

// two f32 -> packed bf16 pair; prefer native converts (v_cvt_pk_bf16_f32)
__device__ __forceinline__ unsigned int pack2bf(float a, float b) {
#if defined(__has_builtin) && __has_builtin(__builtin_amdgcn_cvt_pk_bf16_f32)
  auto pk = __builtin_amdgcn_cvt_pk_bf16_f32(a, b);
  return __builtin_bit_cast(unsigned int, pk);
#else
  union { unsigned int u; __bf16 h[2]; } r;
  r.h[0] = (__bf16)a;   // native fptrunc f32->bf16 on gfx1250
  r.h[1] = (__bf16)b;
  return r.u;
#endif
}

__device__ __forceinline__ void atomicMaxF(float* addr, float v) {
  if (v >= 0.f) atomicMax((int*)addr, __float_as_int(v));
  else          atomicMin((unsigned int*)addr, (unsigned int)__float_as_int(v));
}

__device__ __forceinline__ float gelu_exact(float x) {
  return 0.5f * x * (1.f + erff(x * 0.7071067811865475f));
}

// ---------------------------------------------------------------- utility
__global__ void hgt_fill(float* __restrict__ p, int n, float v) {
  int i = blockIdx.x * 256 + threadIdx.x;
  if (i < n) p[i] = v;
}
__global__ void hgt_ifill(int* __restrict__ p, int n, int v) {
  int i = blockIdx.x * 256 + threadIdx.x;
  if (i < n) p[i] = v;
}

// ---------------------------------------------------------- type sorting
__global__ void hgt_hist(const int* __restrict__ types, int* __restrict__ cnt) {
  int i = blockIdx.x * 256 + threadIdx.x;
  if (i < NN) atomicAdd(&cnt[types[i]], 1);
}
__global__ void hgt_scan(const int* __restrict__ cnt, int* __restrict__ offp) {
  if (threadIdx.x == 0 && blockIdx.x == 0) {
    int o = 0;
    for (int t = 0; t < TT; t++) {
      offp[t] = o;
      o += (cnt[t] + 127) & ~127;   // pad each bucket to 128 rows
    }
    offp[TT] = o;
  }
}
__global__ void hgt_scatter(const int* __restrict__ types, int* __restrict__ cnt2,
                            const int* __restrict__ offp, int* __restrict__ perm) {
  int i = blockIdx.x * 256 + threadIdx.x;
  if (i >= NN) return;
  int t = types[i];
  int p = atomicAdd(&cnt2[t], 1);
  perm[offp[t] + p] = i;
}

// Pack f32 weights [T][KD][128] into WMMA B-fragment order (bf16 pairs).
// idx = (((t*nkt + kt)*8 + ct)*32 + lane)*8 + i
// lane: g=lane>>4, n=lane&15 ; half j of lane = B[kt*32 + 8*g + j + (j>=8 ? 8 : 0)][ct*16+n]
__global__ void hgt_pack(const float* __restrict__ W, unsigned int* __restrict__ out,
                         int nkt, int KD) {
  int idx = blockIdx.x * 256 + threadIdx.x;
  int total = TT * nkt * 8 * 256;
  if (idx >= total) return;
  int i    = idx & 7;
  int lane = (idx >> 3) & 31;
  int ct   = (idx >> 8) & 7;
  int kt   = (idx >> 11) % nkt;
  int t    = idx / (2048 * nkt);
  int g = lane >> 4, n = lane & 15;
  int K   = kt * 32 + 8 * g + 2 * i + ((i >= 4) ? 8 : 0);
  int col = ct * 16 + n;
  const float* Wt = W + (size_t)t * KD * NHID;
  out[idx] = pack2bf(Wt[(size_t)K * NHID + col], Wt[(size_t)(K + 1) * NHID + col]);
}

// rte_vec[p][j] = RTE_TAB[p] @ rte_W[l] + rte_b[l]   (240 x 128)
__global__ void hgt_rte_vec(const float* __restrict__ rW, const float* __restrict__ rb,
                            float* __restrict__ outv) {
  int idx = blockIdx.x * 256 + threadIdx.x;
  if (idx >= MAXLEN * NHID) return;
  int p = idx / NHID, j = idx % NHID;
  float acc = rb[j];
  const float c = -logf(10000.f) / (float)NHID;
  #pragma unroll 4
  for (int i = 0; i < NHID; i++) {
    int m = i >> 1;
    float dv  = __expf((float)(2 * m) * c);
    float arg = (float)p * dv;
    float tv  = ((i & 1) ? __cosf(arg) : __sinf(arg)) * 0.08838834764831845f;
    acc = fmaf(tv, rW[i * NHID + j], acc);
  }
  outv[idx] = acc;
}

// out[p][t][j] = rte_vec[p] @ W[t]   (240 x 3 x 128), no bias
__global__ void hgt_rte_proj(const float* __restrict__ rv, const float* __restrict__ Wl,
                             float* __restrict__ out) {
  int idx = blockIdx.x * 256 + threadIdx.x;
  if (idx >= MAXLEN * TT * NHID) return;
  int j = idx % NHID;
  int t = (idx / NHID) % TT;
  int p = idx / (NHID * TT);
  const float* W = Wl + (size_t)t * NHID * NHID;
  float acc = 0.f;
  #pragma unroll 4
  for (int i = 0; i < NHID; i++) acc = fmaf(rv[p * NHID + i], W[i * NHID + j], acc);
  out[idx] = acc;
}

// --------------------------- WMMA linear over type-sorted node tiles
// Each tile is type-homogeneous (buckets padded); perm[row] = node or -1.
// B fragments staged in LDS (shared by all waves); A fragments built directly
// from global X rows (two 32B chunks per k-step per lane).
// EPI: 0 = plain (+bias), 1 = tanh(+bias), 2 = gelu-input, gated skip + LayerNorm
template <int KD, int EPI>
__global__ __launch_bounds__(256) void hgt_mms(
    const float* __restrict__ X, const unsigned int* __restrict__ PW,
    const float* __restrict__ bias, const int* __restrict__ perm,
    const int* __restrict__ offp, float* __restrict__ Y,
    const float* __restrict__ xold, const float* __restrict__ skipv,
    const float* __restrict__ lng, const float* __restrict__ lnb) {
  constexpr int ROWS = (KD > 128) ? 64 : 128;
  constexpr int NKT  = KD / 32;
  constexpr int NPH  = NKT / 4;            // phases of 4 k-steps (32KB B each)
  __shared__ unsigned int sB[4 * 8 * 256]; // B frags for one phase
  __shared__ int   sPerm[ROWS];
  __shared__ int   sT;
  __shared__ float sBias[NHID];
  __shared__ float sLng[(EPI == 2) ? NHID : 1];
  __shared__ float sLnb[(EPI == 2) ? NHID : 1];

  const int rowbase = blockIdx.x * ROWS;
  if (threadIdx.x == 0) {
    int o1 = offp[1], o2 = offp[2];
    sT = (rowbase < o1) ? 0 : ((rowbase < o2) ? 1 : 2);
  }
  if (threadIdx.x < ROWS) sPerm[threadIdx.x] = perm[rowbase + threadIdx.x];
  __syncthreads();
  const int t = sT;

  const int w = threadIdx.x >> 5;          // wave id, owns 16-row subtile
  const int lane = threadIdx.x & 31;
  const int g = lane >> 4, n = lane & 15;
  const int nodeA = sPerm[w * 16 + n];     // row feeding this lane's A frags
  const float* Xrow = X + (size_t)((nodeA >= 0) ? nodeA : 0) * KD;

  v8f acc[8];
  #pragma unroll
  for (int ct = 0; ct < 8; ct++)
    #pragma unroll
    for (int j = 0; j < 8; j++) acc[ct][j] = 0.f;

  const unsigned int* PWt = PW + (size_t)t * NKT * 8 * 256;
  for (int ph = 0; ph < NPH; ph++) {
    if (ph > 0) __syncthreads();           // protect sB reuse
    {  // cooperative stage: 4 k-steps of B fragments -> LDS (coalesced b128)
      const uint4* srcq = (const uint4*)(PWt + (size_t)ph * 4 * 8 * 256);
      uint4* dstq = (uint4*)sB;
      for (int u = threadIdx.x; u < 4 * 8 * 256 / 4; u += blockDim.x) dstq[u] = srcq[u];
    }
    if (ph == 0) {
      for (int u = threadIdx.x; u < NHID; u += blockDim.x) {
        sBias[u] = bias[t * NHID + u];
        if constexpr (EPI == 2) {
          sLng[u] = lng[t * NHID + u];
          sLnb[u] = lnb[t * NHID + u];
        }
      }
    }
    if (ph + 1 < NPH)
      __builtin_prefetch((const void*)(PWt + (size_t)(ph + 1) * 4 * 8 * 256), 0, 0);
    __syncthreads();

    #pragma unroll
    for (int k4 = 0; k4 < 4; k4++) {
      const int kt = ph * 4 + k4;
      Frag a;
      if (nodeA >= 0) {
        const float4* p0 = (const float4*)(Xrow + kt * 32 + 8 * g);
        const float4* p1 = (const float4*)(Xrow + kt * 32 + 16 + 8 * g);
        float4 d0 = p0[0], d1 = p0[1], d2 = p1[0], d3 = p1[1];
        if constexpr (EPI == 2) {
          d0.x = gelu_exact(d0.x); d0.y = gelu_exact(d0.y); d0.z = gelu_exact(d0.z); d0.w = gelu_exact(d0.w);
          d1.x = gelu_exact(d1.x); d1.y = gelu_exact(d1.y); d1.z = gelu_exact(d1.z); d1.w = gelu_exact(d1.w);
          d2.x = gelu_exact(d2.x); d2.y = gelu_exact(d2.y); d2.z = gelu_exact(d2.z); d2.w = gelu_exact(d2.w);
          d3.x = gelu_exact(d3.x); d3.y = gelu_exact(d3.y); d3.z = gelu_exact(d3.z); d3.w = gelu_exact(d3.w);
        }
        a.u[0] = pack2bf(d0.x, d0.y); a.u[1] = pack2bf(d0.z, d0.w);
        a.u[2] = pack2bf(d1.x, d1.y); a.u[3] = pack2bf(d1.z, d1.w);
        a.u[4] = pack2bf(d2.x, d2.y); a.u[5] = pack2bf(d2.z, d2.w);
        a.u[6] = pack2bf(d3.x, d3.y); a.u[7] = pack2bf(d3.z, d3.w);
      } else {
        #pragma unroll
        for (int i = 0; i < 8; i++) a.u[i] = 0u;
      }
      const uint4* sBq = (const uint4*)sB;
      Frag b[8];
      #pragma unroll
      for (int ct = 0; ct < 8; ct++) {     // batch LDS b128 loads
        b[ct].q[0] = sBq[(k4 * 8 + ct) * 64 + lane * 2];
        b[ct].q[1] = sBq[(k4 * 8 + ct) * 64 + lane * 2 + 1];
      }
      #pragma unroll
      for (int ct = 0; ct < 8; ct++)
        acc[ct] = __builtin_amdgcn_wmma_f32_16x16x32_bf16(
            false, a.v, false, b[ct].v, (short)0, acc[ct], false, false);
    }
  }

  // epilogue (C/D layout: row = i + 8*g, col = ct*16 + n)
  if constexpr (EPI == 2) {
    float al = 1.f / (1.f + __expf(-skipv[t]));
    #pragma unroll
    for (int ct = 0; ct < 8; ct++) {
      int col = ct * 16 + n;
      #pragma unroll
      for (int i = 0; i < 8; i++) {
        int node = sPerm[w * 16 + i + 8 * g];
        float v = acc[ct][i] + sBias[col];
        float xo = (node >= 0) ? xold[(size_t)node * NHID + col] : 0.f;
        acc[ct][i] = al * v + (1.f - al) * xo;
      }
    }
    #pragma unroll
    for (int i = 0; i < 8; i++) {
      float s = 0.f, sq = 0.f;
      #pragma unroll
      for (int ct = 0; ct < 8; ct++) { float v = acc[ct][i]; s += v; sq += v * v; }
      for (int m = 1; m < 16; m <<= 1) {
        s  += __shfl_xor(s,  m, 32);
        sq += __shfl_xor(sq, m, 32);
      }
      float mu  = s * (1.f / 128.f);
      float var = sq * (1.f / 128.f) - mu * mu;
      float rv  = rsqrtf(var + 1e-5f);
      int node = sPerm[w * 16 + i + 8 * g];
      if (node >= 0) {
        #pragma unroll
        for (int ct = 0; ct < 8; ct++) {
          int col = ct * 16 + n;
          Y[(size_t)node * NHID + col] =
              (acc[ct][i] - mu) * rv * sLng[col] + sLnb[col];
        }
      }
    }
  } else {
    #pragma unroll
    for (int i = 0; i < 8; i++) {
      int node = sPerm[w * 16 + i + 8 * g];
      if (node >= 0) {
        #pragma unroll
        for (int ct = 0; ct < 8; ct++) {
          float v = acc[ct][i] + sBias[ct * 16 + n];
          if constexpr (EPI == 1) v = tanhf(v);
          Y[(size_t)node * NHID + ct * 16 + n] = v;
        }
      }
    }
  }
}

// --------------------------------------------------------- edge kernels
__global__ __launch_bounds__(256) void hgt_score(
    const float* __restrict__ K0, const float* __restrict__ Q0,
    const float* __restrict__ RK, const float* __restrict__ relA,
    const float* __restrict__ relP, const int* __restrict__ ntype,
    const int* __restrict__ eidx, const int* __restrict__ etime,
    const int* __restrict__ etype, float* __restrict__ score,
    float* __restrict__ smax) {
  __shared__ float sA[RR * 4 * 256];
  const int hg = blockIdx.y;
  for (int u = threadIdx.x; u < RR * 4 * 256; u += 256) {
    int r = u >> 10, hh = (u >> 8) & 3, dd = u & 255;
    sA[u] = relA[((r * HH) + (hg * 4 + hh)) * 256 + dd];
  }
  __syncthreads();
  int e = blockIdx.x * 64 + (threadIdx.x >> 2);
  if (e >= EE) return;
  int h = hg * 4 + (threadIdx.x & 3);
  int src = eidx[e], dst = eidx[EE + e];
  int r = etype[e], tm = etime[e], ts = ntype[src];
  const float4* kp = (const float4*)(K0 + (size_t)src * NHID + h * DKK);
  const float4* rp = (const float4*)(RK + ((size_t)tm * TT + ts) * NHID + h * DKK);
  const float4* qp = (const float4*)(Q0 + (size_t)dst * NHID + h * DKK);
  float k[16], q[16];
  #pragma unroll
  for (int i = 0; i < 4; i++) {
    float4 a = kp[i], b = rp[i], c = qp[i];
    k[4*i] = a.x + b.x; k[4*i+1] = a.y + b.y; k[4*i+2] = a.z + b.z; k[4*i+3] = a.w + b.w;
    q[4*i] = c.x; q[4*i+1] = c.y; q[4*i+2] = c.z; q[4*i+3] = c.w;
  }
  const float* A = sA + ((r << 2) + (h & 3)) * 256;
  float s = 0.f;
  #pragma unroll
  for (int dk = 0; dk < 16; dk++) {
    float t = 0.f;
    #pragma unroll
    for (int d = 0; d < 16; d++) t = fmaf(k[d], A[d * 16 + dk], t);
    s = fmaf(q[dk], t, s);
  }
  s *= relP[r * HH + h] * 0.25f;
  score[(size_t)e * HH + h] = s;
  atomicMaxF(&smax[(size_t)dst * HH + h], s);
}

__global__ void hgt_expnorm(float* __restrict__ score, const float* __restrict__ smax,
                            float* __restrict__ denom, const int* __restrict__ eidx) {
  int gid = blockIdx.x * 256 + threadIdx.x;
  if (gid >= EE * HH) return;
  int e = gid >> 3, h = gid & 7;
  int dst = eidx[EE + e];
  float v = __expf(score[gid] - smax[(size_t)dst * HH + h]);
  score[gid] = v;
  atomicAdd(&denom[(size_t)dst * HH + h], v);
}

__global__ __launch_bounds__(256) void hgt_aggr(
    const float* __restrict__ V0, const float* __restrict__ RV,
    const float* __restrict__ relM, const float* __restrict__ esc,
    const float* __restrict__ denom, const int* __restrict__ ntype,
    const int* __restrict__ eidx, const int* __restrict__ etime,
    const int* __restrict__ etype, float* __restrict__ aggr) {
  __shared__ float sM[RR * 4 * 256];
  const int hg = blockIdx.y;
  for (int u = threadIdx.x; u < RR * 4 * 256; u += 256) {
    int r = u >> 10, hh = (u >> 8) & 3, dd = u & 255;
    sM[u] = relM[((r * HH) + (hg * 4 + hh)) * 256 + dd];
  }
  __syncthreads();
  int e = blockIdx.x * 64 + (threadIdx.x >> 2);
  if (e >= EE) return;
  int h = hg * 4 + (threadIdx.x & 3);
  int src = eidx[e], dst = eidx[EE + e];
  int r = etype[e], tm = etime[e], ts = ntype[src];
  float att = esc[(size_t)e * HH + h] / (denom[(size_t)dst * HH + h] + 1e-16f);
  const float4* vp = (const float4*)(V0 + (size_t)src * NHID + h * DKK);
  const float4* rp = (const float4*)(RV + ((size_t)tm * TT + ts) * NHID + h * DKK);
  float v[16];
  #pragma unroll
  for (int i = 0; i < 4; i++) {
    float4 a = vp[i], b = rp[i];
    v[4*i] = a.x + b.x; v[4*i+1] = a.y + b.y; v[4*i+2] = a.z + b.z; v[4*i+3] = a.w + b.w;
  }
  const float* M = sM + ((r << 2) + (h & 3)) * 256;
  float* out = aggr + (size_t)dst * NHID + h * DKK;
  #pragma unroll
  for (int kk = 0; kk < 16; kk++) {
    float m = 0.f;
    #pragma unroll
    for (int d = 0; d < 16; d++) m = fmaf(v[d], M[d * 16 + kk], m);
    atomicAdd(&out[kk], m * att);
  }
}

// ------------------------------------------------------------------ host
extern "C" void kernel_launch(void* const* d_in, const int* in_sizes, int n_in,
                              void* d_out, int out_size, void* d_ws, size_t ws_size,
                              hipStream_t stream) {
  const float* node_feature = (const float*)d_in[0];
  const float* adapt_W = (const float*)d_in[1];
  const float* adapt_b = (const float*)d_in[2];
  const float* Wk = (const float*)d_in[3];
  const float* bk = (const float*)d_in[4];
  const float* Wq = (const float*)d_in[5];
  const float* bq = (const float*)d_in[6];
  const float* Wv = (const float*)d_in[7];
  const float* bv = (const float*)d_in[8];
  const float* Wa = (const float*)d_in[9];
  const float* ba = (const float*)d_in[10];
  const float* rel_pri = (const float*)d_in[11];
  const float* rel_att = (const float*)d_in[12];
  const float* rel_msg = (const float*)d_in[13];
  const float* skip = (const float*)d_in[14];
  const float* ln_g = (const float*)d_in[15];
  const float* ln_b = (const float*)d_in[16];
  const float* rte_W = (const float*)d_in[17];
  const float* rte_b = (const float*)d_in[18];
  const int* node_type = (const int*)d_in[19];
  const int* edge_time = (const int*)d_in[20];
  const int* edge_index = (const int*)d_in[21];
  const int* edge_type = (const int*)d_in[22];

  char* base = (char*)d_ws;
  size_t off = 0;
  auto carve = [&](size_t bytes) -> void* {
    void* p = base + off;
    off = (off + bytes + 255) & ~(size_t)255;
    return p;
  };
  float* x0    = (float*)carve((size_t)NN * NHID * 4);
  float* x1    = (float*)carve((size_t)NN * NHID * 4);
  float* Q0    = (float*)carve((size_t)NN * NHID * 4);
  float* K0    = (float*)carve((size_t)NN * NHID * 4);
  float* V0    = (float*)carve((size_t)NN * NHID * 4);
  float* aggr  = (float*)carve((size_t)NN * NHID * 4);
  float* smax  = (float*)carve((size_t)NN * HH * 4);
  float* denom = (float*)carve((size_t)NN * HH * 4);
  float* esc   = (float*)carve((size_t)EE * HH * 4);
  float* rtev  = (float*)carve((size_t)MAXLEN * NHID * 4);
  float* RK    = (float*)carve((size_t)MAXLEN * TT * NHID * 4);
  float* RV    = (float*)carve((size_t)MAXLEN * TT * NHID * 4);
  int*   cnt   = (int*)carve(4 * sizeof(int));
  int*   cnt2  = (int*)carve(4 * sizeof(int));
  int*   offp  = (int*)carve(4 * sizeof(int));
  int*   perm  = (int*)carve((size_t)PAD_CAP * sizeof(int));
  unsigned int* pAdapt = (unsigned int*)carve((size_t)TT * 8 * 2048 * 4);
  unsigned int* pW[LL][4];
  for (int l = 0; l < LL; l++)
    for (int m = 0; m < 4; m++)
      pW[l][m] = (unsigned int*)carve((size_t)TT * 4 * 2048 * 4);

  // ---- counting sort of nodes by type (padded to 128-row tiles)
  hgt_ifill<<<1, 64, 0, stream>>>(cnt, 4, 0);
  hgt_ifill<<<1, 64, 0, stream>>>(cnt2, 4, 0);
  hgt_ifill<<<(PAD_CAP + 255) / 256, 256, 0, stream>>>(perm, PAD_CAP, -1);
  hgt_hist<<<(NN + 255) / 256, 256, 0, stream>>>(node_type, cnt);
  hgt_scan<<<1, 1, 0, stream>>>(cnt, offp);
  hgt_scatter<<<(NN + 255) / 256, 256, 0, stream>>>(node_type, cnt2, offp, perm);

  // ---- pack all weights into WMMA B-fragment layout (bf16)
  hgt_pack<<<(TT * 8 * 2048 + 255) / 256, 256, 0, stream>>>(adapt_W, pAdapt, 8, INDIM);
  for (int l = 0; l < LL; l++) {
    const float* Ws[4] = {Wq + (size_t)l * TT * NHID * NHID, Wk + (size_t)l * TT * NHID * NHID,
                          Wv + (size_t)l * TT * NHID * NHID, Wa + (size_t)l * TT * NHID * NHID};
    for (int m = 0; m < 4; m++)
      hgt_pack<<<(TT * 4 * 2048 + 255) / 256, 256, 0, stream>>>(Ws[m], pW[l][m], 4, NHID);
  }

  // ---- adapt: x0 = tanh(node_feature @ adapt_W[type] + adapt_b[type])
  hgt_mms<INDIM, 1><<<PAD_CAP / 64, 128, 0, stream>>>(
      node_feature, pAdapt, adapt_b, perm, offp, x0,
      nullptr, nullptr, nullptr, nullptr);

  float* xcur = x0;
  float* xnext = x1;
  const float NEGINF = -__builtin_huge_valf();

  for (int l = 0; l < LL; l++) {
    // RTE: 240-row table through rte_W, then per-type Wk / Wv
    hgt_rte_vec<<<(MAXLEN * NHID + 255) / 256, 256, 0, stream>>>(
        rte_W + (size_t)l * NHID * NHID, rte_b + (size_t)l * NHID, rtev);
    hgt_rte_proj<<<(MAXLEN * TT * NHID + 255) / 256, 256, 0, stream>>>(
        rtev, Wk + (size_t)l * TT * NHID * NHID, RK);
    hgt_rte_proj<<<(MAXLEN * TT * NHID + 255) / 256, 256, 0, stream>>>(
        rtev, Wv + (size_t)l * TT * NHID * NHID, RV);

    // node-level projections (WMMA over type-sorted tiles)
    hgt_mms<NHID, 0><<<PAD_CAP / 128, 256, 0, stream>>>(
        xcur, pW[l][0], bq + (size_t)l * TT * NHID, perm, offp, Q0,
        nullptr, nullptr, nullptr, nullptr);
    hgt_mms<NHID, 0><<<PAD_CAP / 128, 256, 0, stream>>>(
        xcur, pW[l][1], bk + (size_t)l * TT * NHID, perm, offp, K0,
        nullptr, nullptr, nullptr, nullptr);
    hgt_mms<NHID, 0><<<PAD_CAP / 128, 256, 0, stream>>>(
        xcur, pW[l][2], bv + (size_t)l * TT * NHID, perm, offp, V0,
        nullptr, nullptr, nullptr, nullptr);

    // segment softmax buffers
    hgt_fill<<<(NN * HH + 255) / 256, 256, 0, stream>>>(smax, NN * HH, NEGINF);
    hgt_fill<<<(NN * HH + 255) / 256, 256, 0, stream>>>(denom, NN * HH, 0.f);
    hgt_fill<<<(NN * NHID + 255) / 256, 256, 0, stream>>>(aggr, NN * NHID, 0.f);

    dim3 egrid((EE + 63) / 64, 2);
    hgt_score<<<egrid, 256, 0, stream>>>(
        K0, Q0, RK, rel_att + (size_t)l * RR * HH * 256, rel_pri + (size_t)l * RR * HH,
        node_type, edge_index, edge_time, edge_type, esc, smax);
    hgt_expnorm<<<(EE * HH + 255) / 256, 256, 0, stream>>>(esc, smax, denom, edge_index);
    hgt_aggr<<<egrid, 256, 0, stream>>>(
        V0, RV, rel_msg + (size_t)l * RR * HH * 256, esc, denom,
        node_type, edge_index, edge_time, edge_type, aggr);

    // update: gelu -> Wa[type] -> gated skip -> LayerNorm (WMMA, sorted tiles)
    float* outp = (l == LL - 1) ? (float*)d_out : xnext;
    hgt_mms<NHID, 2><<<PAD_CAP / 128, 256, 0, stream>>>(
        aggr, pW[l][3], ba + (size_t)l * TT * NHID, perm, offp, outp,
        xcur, skip + (size_t)l * TT, ln_g + (size_t)l * TT * NHID,
        ln_b + (size_t)l * TT * NHID);

    float* tmp = xcur; xcur = xnext; xnext = tmp;
  }
}
